// WhisperLogMel_30090540876226
// MI455X (gfx1250) — matrix operand order repaired and checked by reading, not compiled, not run on hardware
//
#include <hip/hip_runtime.h>
#include <hip/hip_bf16.h>

// ---------------- types ----------------
typedef _Float16 v16h __attribute__((ext_vector_type(16)));
typedef _Float16 v8h  __attribute__((ext_vector_type(8)));
typedef float    v8f  __attribute__((ext_vector_type(8)));

// ---------------- problem constants ----------------
#define BATCH    32
#define CHUNK    480000
#define NFFT     400
#define HOP      160
#define NFREQ    201
#define NMELS    128
#define NFRAMES  3000

// GEMM shapes
#define KS_DFT   13              // 400 -> 13 k-steps of 32 (padded 416)
#define MT_DFT   13              // 208 freq rows / 16
#define KS_MEL   7               // 201 -> 7 k-steps of 32 (padded 224)
#define MT_MEL   8               // 128 mel rows / 16
#define KP_MEL   224

// workgroup tiling: 64 frames = 4 n-tiles, accumulators held in registers
#define FR_WG    64
#define NT_WG    4
#define NGROUPS  47              // ceil(3000/64)

// LDS strides (halves) — strides give conflict-free ds_load_b128 across lanes
#define LDS_FR_STRIDE  424       // >= 416
#define LDS_PW_STRIDE  232       // >= 224

// workspace layout (halves). A matrices stored FRAGMENT-PACKED:
//   DFT: [mt 13][ks 13][cs 2][lane 32][j 16]   (cs: 0=cos, 1=sin)
//   MEL: [mt 8 ][ks 7 ]      [lane 32][j 16]
#define SZ_A_DFT  (MT_DFT * KS_DFT * 2 * 512)   // 173056 halves
#define OFF_MEL   SZ_A_DFT
#define SZ_A_MEL  (MT_MEL * KS_MEL * 512)       // 28672 halves
#define OFF_MAX_B ((OFF_MEL + SZ_A_MEL) * 2)    // byte offset of u32 max array
#define PREP_N    (OFF_MEL + SZ_A_MEL + BATCH)

// ---------------- helpers ----------------
__device__ __forceinline__ unsigned ord_map(float f) {
    unsigned u = __float_as_uint(f);
    return (u & 0x80000000u) ? ~u : (u | 0x80000000u);
}
__device__ __forceinline__ float ord_unmap(unsigned u) {
    return (u & 0x80000000u) ? __uint_as_float(u & 0x7fffffffu)
                             : __uint_as_float(~u);
}

// B fragment (32x16, source stored [N][K] row-major): lane -> N = lane&15,
// half j -> K = c + j,  c = ks*32 + (lane>=16 ? 16 : 0)
__device__ __forceinline__ v16h load_frag_b(const _Float16* rowptr, int c) {
    v8h lo = *(const v8h*)(rowptr + c);
    v8h hi = *(const v8h*)(rowptr + c + 8);
    v16h r;
#pragma unroll
    for (int j = 0; j < 8; ++j) { r[j] = lo[j]; r[j + 8] = hi[j]; }
    return r;
}

// ---------------- kernel 1: pack A matrices (fragment-swizzled f16) ----------------
__global__ void wlm_prep(const float* __restrict__ cosk,
                         const float* __restrict__ sink,
                         const float* __restrict__ melf,
                         _Float16* __restrict__ wsh,
                         unsigned* __restrict__ wsmax) {
    for (int i = blockIdx.x * blockDim.x + threadIdx.x; i < PREP_N;
         i += gridDim.x * blockDim.x) {
        if (i < SZ_A_DFT) {                       // DFT A (cos/sin interleaved)
            int j    = i & 15;
            int lane = (i >> 4) & 31;
            int cs   = (i >> 9) & 1;
            int km   = i >> 10;
            int ks   = km % KS_DFT, mt = km / KS_DFT;
            int M    = mt * 16 + (lane & 15);
            int K    = ks * 32 + ((lane >> 4) ? 16 : 0) + j;
            float v  = 0.f;
            if (M < NFREQ && K < NFFT)
                v = (cs ? sink : cosk)[M * NFFT + K];
            wsh[i] = (_Float16)v;
        } else if (i < OFF_MEL + SZ_A_MEL) {      // MEL A
            int p    = i - OFF_MEL;
            int j    = p & 15;
            int lane = (p >> 4) & 31;
            int km   = p >> 9;
            int ks   = km % KS_MEL, mt = km / KS_MEL;
            int M    = mt * 16 + (lane & 15);     // < 128 always
            int K    = ks * 32 + ((lane >> 4) ? 16 : 0) + j;
            float v  = (K < NFREQ) ? melf[M * NFREQ + K] : 0.f;
            wsh[i] = (_Float16)v;
        } else {                                  // per-batch ordered-max init
            wsmax[i - (OFF_MEL + SZ_A_MEL)] = 0u;
        }
    }
}

// ---------------- kernel 2: fused DFT(+power)+mel per 64-frame group ----------------
__global__ __launch_bounds__(256)
void wlm_main(const float* __restrict__ audio,
              const _Float16* __restrict__ wsh,
              unsigned* __restrict__ wsmax,
              float* __restrict__ out) {
    __shared__ _Float16 lds_fr[FR_WG * LDS_FR_STRIDE];   // framed audio, f16
    __shared__ _Float16 lds_pw[FR_WG * LDS_PW_STRIDE];   // power spectrum, f16

    const int b    = blockIdx.y;
    const int t0   = blockIdx.x * FR_WG;
    const int tid  = threadIdx.x;
    const int lane = tid & 31;
    const int wave = tid >> 5;
    const int hi   = (lane >> 4) & 1;
    const int l16  = lane & 15;

    // ---- stage: reflect-padded overlapping frames -> LDS f16; zero power tile ----
    const float* ab = audio + (size_t)b * CHUNK;
    for (int i = tid; i < FR_WG * LDS_FR_STRIDE; i += 256) {
        int fr = i / LDS_FR_STRIDE, k = i % LDS_FR_STRIDE;
        int t = t0 + fr;
        float v = 0.f;
        if (t < NFRAMES && k < NFFT) {
            int p = t * HOP + k - 200;                // reflect pad by 200
            if (p < 0) p = -p;
            if (p >= CHUNK) p = 2 * CHUNK - 2 - p;
            v = ab[p];
        }
        lds_fr[i] = (_Float16)v;
    }
    for (int i = tid; i < FR_WG * LDS_PW_STRIDE; i += 256)
        lds_pw[i] = (_Float16)0.f;
    __syncthreads();

    // ---- phase 1: DFT GEMM. Each wave owns an M-tile, 4 N-tiles in registers ----
    for (int mt = wave; mt < MT_DFT; mt += 8) {
        v8f accR[NT_WG] = {}, accI[NT_WG] = {};
        const _Float16* pa = wsh + ((size_t)mt * KS_DFT * 2 * 512) + lane * 16;
#pragma unroll
        for (int ks = 0; ks < KS_DFT; ++ks) {
            v16h ac = *(const v16h*)(pa);            // coalesced 1KB fragment
            v16h as = *(const v16h*)(pa + 512);
            pa += 1024;
            int cb = ks * 32 + (hi ? 16 : 0);
#pragma unroll
            for (int nt = 0; nt < NT_WG; ++nt) {
                v16h bb = load_frag_b(&lds_fr[(nt * 16 + l16) * LDS_FR_STRIDE], cb);
                accR[nt] = __builtin_amdgcn_wmma_f32_16x16x32_f16(
                    false, ac, false, bb, (short)0, accR[nt], false, false);
                accI[nt] = __builtin_amdgcn_wmma_f32_16x16x32_f16(
                    false, as, false, bb, (short)0, accI[nt], false, false);
            }
        }
#pragma unroll
        for (int nt = 0; nt < NT_WG; ++nt) {
            int fr = nt * 16 + l16;
#pragma unroll
            for (int r = 0; r < 8; ++r) {
                int f = mt * 16 + r + (hi ? 8 : 0);   // <= 207 < KP_MEL
                float p = accR[nt][r] * accR[nt][r] + accI[nt][r] * accI[nt][r];
                lds_pw[fr * LDS_PW_STRIDE + f] = (_Float16)fminf(p, 60000.f);
            }
        }
    }
    __syncthreads();

    // ---- phase 2: mel GEMM (wave == M-tile, perfectly balanced) + epilogue ----
    float wmax = -INFINITY;
    {
        const int mt = wave;                          // 8 waves, 8 mel M-tiles
        v8f acc[NT_WG] = {};
        const _Float16* pa = wsh + OFF_MEL + ((size_t)mt * KS_MEL * 512) + lane * 16;
#pragma unroll
        for (int ks = 0; ks < KS_MEL; ++ks) {
            v16h aa = *(const v16h*)(pa);
            pa += 512;
            int cb = ks * 32 + (hi ? 16 : 0);
#pragma unroll
            for (int nt = 0; nt < NT_WG; ++nt) {
                v16h bb = load_frag_b(&lds_pw[(nt * 16 + l16) * LDS_PW_STRIDE], cb);
                acc[nt] = __builtin_amdgcn_wmma_f32_16x16x32_f16(
                    false, aa, false, bb, (short)0, acc[nt], false, false);
            }
        }
#pragma unroll
        for (int nt = 0; nt < NT_WG; ++nt) {
            int t = t0 + nt * 16 + l16;
            if (t < NFRAMES) {
                size_t base = (size_t)b * NMELS * NFRAMES + t;
#pragma unroll
                for (int r = 0; r < 8; ++r) {
                    int m = mt * 16 + r + (hi ? 8 : 0);
                    float lm = log10f(fmaxf(acc[nt][r], 1e-10f));
                    out[base + (size_t)m * NFRAMES] = lm;
                    wmax = fmaxf(wmax, lm);
                }
            }
        }
    }
    // wave32 max reduction, one ordered-uint atomic per wave
#pragma unroll
    for (int off = 16; off; off >>= 1)
        wmax = fmaxf(wmax, __shfl_xor(wmax, off, 32));
    if (lane == 0) atomicMax(&wsmax[b], ord_map(wmax));
}

// ---------------- kernel 3: clamp to (max-8), scale, in place ----------------
__global__ void wlm_final(float* __restrict__ out,
                          const unsigned* __restrict__ wsmax) {
    int i = blockIdx.x * blockDim.x + threadIdx.x;   // grid sized exactly
    int b = i / (NMELS * NFRAMES);
    float mx = ord_unmap(wsmax[b]);
    float v = out[i];
    v = fmaxf(v, mx - 8.0f);
    out[i] = (v + 4.0f) * 0.25f;
}

// ---------------- launcher ----------------
extern "C" void kernel_launch(void* const* d_in, const int* in_sizes, int n_in,
                              void* d_out, int out_size, void* d_ws, size_t ws_size,
                              hipStream_t stream) {
    const float* audio = (const float*)d_in[0];
    const float* cosk  = (const float*)d_in[1];
    const float* sink  = (const float*)d_in[2];
    const float* melf  = (const float*)d_in[3];
    float* out = (float*)d_out;

    _Float16* wsh   = (_Float16*)d_ws;
    unsigned* wsmax = (unsigned*)((char*)d_ws + OFF_MAX_B);

    // 1) pack (fragment-swizzle) A matrices + init per-batch max
    wlm_prep<<<(PREP_N + 255) / 256, 256, 0, stream>>>(cosk, sink, melf, wsh, wsmax);

    // 2) fused DFT + power + mel + log10 + max
    dim3 grid(NGROUPS, BATCH);
    wlm_main<<<grid, 256, 0, stream>>>(audio, wsh, wsmax, out);

    // 3) finalize (exactly one thread per output element)
    int total = BATCH * NMELS * NFRAMES;           // 12,288,000 = 48000*256
    wlm_final<<<total / 256, 256, 0, stream>>>(out, wsmax);
}
